// QuakeWaveMamba2_87909390614598
// MI455X (gfx1250) — compile-verified, matrix operand order checked by reading, not run on hardware
//
#include <hip/hip_runtime.h>
#include <hip/hip_bf16.h>
#include <math.h>
#include <stdint.h>

// ---------------------------------------------------------------------------
// Model dims
// ---------------------------------------------------------------------------
#define BATCH     8
#define SEQ_IN    8192
#define L1_LEN    2048      // after conv1 (stride 4)
#define L_SEQ     1024      // after conv2 (stride 2)
#define D_MODEL   256
#define D_STATE   64
#define D_CONV    4
#define HEADDIM   64
#define N_LAYERS  8
#define D_INNER   1024
#define NHEADS    16
#define CONV_DIM  1152      // D_INNER + 2*D_STATE
#define D_IN_PROJ 2192      // 2*D_INNER + 2*D_STATE + NHEADS
#define ROWS      (BATCH * L_SEQ)   // 8192

typedef __bf16 bf16_t;
typedef bf16_t v16bf __attribute__((ext_vector_type(16)));
typedef bf16_t v8bf  __attribute__((ext_vector_type(8)));
typedef float  v8f   __attribute__((ext_vector_type(8)));

// ---------------------------------------------------------------------------
// Device helpers
// ---------------------------------------------------------------------------
__device__ __forceinline__ float gelu_exact(float x) {
    return 0.5f * x * (1.0f + erff(x * 0.70710678118654752f));
}
__device__ __forceinline__ float silu(float x) {
    return x / (1.0f + __expf(-x));
}
__device__ __forceinline__ float softplus_f(float x) {
    return (x > 20.0f) ? x : log1pf(__expf(x));
}

__device__ __forceinline__ float block_reduce_sum(float v, float* red) {
    int lane = threadIdx.x & 31;
    int wid  = threadIdx.x >> 5;
#pragma unroll
    for (int o = 16; o > 0; o >>= 1) v += __shfl_down(v, o);
    if (lane == 0) red[wid] = v;
    __syncthreads();
    if (threadIdx.x == 0) {
        float s = 0.f;
        int nw = blockDim.x >> 5;
        for (int i = 0; i < nw; ++i) s += red[i];
        red[0] = s;
    }
    __syncthreads();
    float r = red[0];
    __syncthreads();
    return r;
}

// ---------------------------------------------------------------------------
// WMMA bf16 fragment loader (ISA 7.12.2 16-bit A 16x32 layout):
//   lanes 0-15  hold K = {0..7, 16..23};  lanes 16-31 hold K = {8..15, 24..31}
// Works for both A (lane = M row of activations) and B (lane = N row of the
// N x K row-major weight matrix, i.e. B^T column-major).
// ---------------------------------------------------------------------------
__device__ __forceinline__ v16bf load_frag_bf16(const bf16_t* __restrict__ base,
                                                int stride, int row, int kk, int lane) {
    int lo = (lane >> 4) << 3;                 // 0 or 8
    const bf16_t* p0 = base + (size_t)row * stride + kk + lo;
    union { v16bf v; v8bf h[2]; } u;
    u.h[0] = *(const v8bf*)(p0);
    u.h[1] = *(const v8bf*)(p0 + 16);
    return u.v;
}

// ---------------------------------------------------------------------------
// GEMM: D[M][N](f32, ld=N) = A[M][K](bf16) * W[N][K]^T (bf16)
// One wave -> 16(M) x (16*NSUB)(N) strip: NSUB accumulators share one A
// fragment per K-step. NSUB is a compile-time constant so the inner loop is
// straight-line (no EXEC predication around WMMA, no accumulator spills).
// Tail columns (N % (16*NSUB)) are covered by a second launch with NSUB=1
// and an n_start offset. Optional fused bf16 copy of D via template flag.
// ---------------------------------------------------------------------------
template <int NSUB, bool WRITE_BF>
__global__ void __launch_bounds__(256)
gemm_bf16_wmma(const bf16_t* __restrict__ A,
               const bf16_t* __restrict__ W,
               float* __restrict__ D,
               bf16_t* __restrict__ Dbf,
               int M, int N, int K, int n_start, int ngrp) {
    int wave = blockIdx.x * (blockDim.x >> 5) + (threadIdx.x >> 5);
    int lane = threadIdx.x & 31;
    int tile_m = wave / ngrp;
    int grp_n  = wave - tile_m * ngrp;
    if ((tile_m << 4) >= M) return;            // wave-uniform (exact grids)
    int n_base = n_start + grp_n * (NSUB * 16);

    int m  = (tile_m << 4) + (lane & 15);      // A fragment row
    int nr = n_base + (lane & 15);             // W fragment base row

    v8f acc[NSUB] = {};
    for (int kk = 0; kk < K; kk += 32) {
        if (kk + 32 < K)
            __builtin_prefetch(A + (size_t)m * K + kk + 32, 0, 3);
        v16bf a = load_frag_bf16(A, K, m, kk, lane);
#pragma unroll
        for (int j = 0; j < NSUB; ++j) {
            v16bf b = load_frag_bf16(W, K, nr + (j << 4), kk, lane);
            acc[j] = __builtin_amdgcn_wmma_f32_16x16x32_bf16(
                         false, a, false, b, (short)0, acc[j], false, false);
        }
    }

    // C/D layout: VGPR r -> M = r + 8*(lane>=16); N = lane&15
    int row0 = (tile_m << 4) + ((lane >> 4) << 3);
#pragma unroll
    for (int j = 0; j < NSUB; ++j) {
        int col = n_base + (j << 4) + (lane & 15);
#pragma unroll
        for (int r = 0; r < 8; ++r) {
            float v = acc[j][r];
            size_t idx = (size_t)(row0 + r) * N + col;
            D[idx] = v;
            if (WRITE_BF) Dbf[idx] = (bf16_t)v;
        }
    }
}

// ---------------------------------------------------------------------------
// Conv stem
// ---------------------------------------------------------------------------
__global__ void conv1_kernel(const float* __restrict__ x,
                             const float* __restrict__ w,
                             const float* __restrict__ bias,
                             float* __restrict__ out) {
    int idx = blockIdx.x * blockDim.x + threadIdx.x;
    if (idx >= BATCH * 64 * L1_LEN) return;
    int t  = idx & (L1_LEN - 1);
    int co = (idx >> 11) & 63;
    int b  = idx >> 17;
    float acc = bias[co];
    int base = 4 * t - 6;
#pragma unroll
    for (int ci = 0; ci < 3; ++ci) {
        const float* xp = x + ((size_t)b * 3 + ci) * SEQ_IN;
        const float* wp = w + ((size_t)co * 3 + ci) * 16;
#pragma unroll
        for (int k = 0; k < 16; ++k) {
            int s = base + k;
            if (s >= 0 && s < SEQ_IN) acc += xp[s] * wp[k];
        }
    }
    out[idx] = gelu_exact(acc);
}

__global__ void conv2_kernel(const float* __restrict__ h1,
                             const float* __restrict__ w,
                             const float* __restrict__ bias,
                             float* __restrict__ hf,
                             bf16_t* __restrict__ hb) {
    int idx = blockIdx.x * blockDim.x + threadIdx.x;
    if (idx >= BATCH * D_MODEL * L_SEQ) return;
    int t  = idx & (L_SEQ - 1);
    int co = (idx >> 10) & (D_MODEL - 1);
    int b  = idx >> 18;
    float acc = bias[co];
    int base = 2 * t - 7;
    for (int ci = 0; ci < 64; ++ci) {
        const float* hp = h1 + ((size_t)b * 64 + ci) * L1_LEN;
        const float* wp = w + ((size_t)co * 64 + ci) * 16;
#pragma unroll
        for (int k = 0; k < 16; ++k) {
            int s = base + k;
            if (s >= 0 && s < L1_LEN) acc += hp[s] * wp[k];
        }
    }
    float g = gelu_exact(acc);
    size_t o = ((size_t)b * L_SEQ + t) * D_MODEL + co;   // (b, t, c) layout
    hf[o] = g;
    hb[o] = (bf16_t)g;
}

// ---------------------------------------------------------------------------
// dt = softplus(raw + dt_bias); dA = exp(-exp(A_log)*dt)
// ---------------------------------------------------------------------------
__global__ void dtda_kernel(const float* __restrict__ zx,
                            const float* __restrict__ dt_bias,
                            const float* __restrict__ A_log,
                            float* __restrict__ dt_out,
                            float* __restrict__ dA_out, int layer) {
    int idx = blockIdx.x * blockDim.x + threadIdx.x;
    if (idx >= ROWS * NHEADS) return;
    int hd  = idx & (NHEADS - 1);
    int row = idx >> 4;
    float v  = zx[(size_t)row * D_IN_PROJ + (2 * D_INNER + 2 * D_STATE) + hd]
             + dt_bias[layer * NHEADS + hd];
    float dtv = softplus_f(v);
    float a   = -__expf(A_log[layer * NHEADS + hd]);
    dt_out[idx] = dtv;
    dA_out[idx] = __expf(a * dtv);
}

// ---------------------------------------------------------------------------
// Depthwise causal conv (k=4) over seq + SiLU
// ---------------------------------------------------------------------------
__global__ void dwconv_kernel(const float* __restrict__ zx,
                              const float* __restrict__ w,
                              const float* __restrict__ bias,
                              float* __restrict__ xBC, int layer) {
    int idx = blockIdx.x * blockDim.x + threadIdx.x;
    if (idx >= ROWS * CONV_DIM) return;
    int c   = idx % CONV_DIM;
    int row = idx / CONV_DIM;
    int t   = row & (L_SEQ - 1);
    const float* wp = w + ((size_t)layer * CONV_DIM + c) * D_CONV;
    float acc = bias[(size_t)layer * CONV_DIM + c];
#pragma unroll
    for (int k = 0; k < D_CONV; ++k) {
        int tt = t + k - (D_CONV - 1);
        if (tt >= 0)
            acc += zx[(size_t)(row + k - (D_CONV - 1)) * D_IN_PROJ + D_INNER + c] * wp[k];
    }
    xBC[idx] = silu(acc);
}

// ---------------------------------------------------------------------------
// Sequential SSM scan. One 64-thread (2-wave) workgroup per (batch, head).
// Lane p owns state row h[p][0..63] in VGPRs. B_t / C_t are broadcast via
// LDS using CDNA5 async global->LDS loads (ASYNCcnt-tracked, no VGPR bounce).
// ---------------------------------------------------------------------------
__global__ void __launch_bounds__(64)
scan_kernel(const float* __restrict__ xBC,
            const float* __restrict__ dt,
            const float* __restrict__ dA,
            const float* __restrict__ ssm_D,
            float* __restrict__ y, int layer) {
    int bh = blockIdx.x;
    int b  = bh >> 4;
    int hd = bh & 15;
    int p  = threadIdx.x;            // 0..63 = HEADDIM index

    __shared__ float Bs[D_STATE];
    __shared__ float Cs[D_STATE];

    // Wave-relative LDS byte offsets (flat LDS pointer low bits, ISA §10.2)
    unsigned ldsB = (unsigned)(uintptr_t)(&Bs[p]);
    unsigned ldsC = (unsigned)(uintptr_t)(&Cs[p]);

    float st[D_STATE];
#pragma unroll
    for (int n = 0; n < D_STATE; ++n) st[n] = 0.f;

    const float Dv = ssm_D[layer * NHEADS + hd];

    for (int t = 0; t < L_SEQ; ++t) {
        int row = b * L_SEQ + t;
        const float* xb = xBC + (size_t)row * CONV_DIM;

        // async memory -> LDS broadcast of B_t, C_t (one b32 per lane)
        unsigned long long gB = (unsigned long long)(xb + D_INNER + p);
        unsigned long long gC = (unsigned long long)(xb + D_INNER + D_STATE + p);
        asm volatile(
            "global_load_async_to_lds_b32 %0, %2, off\n\t"
            "global_load_async_to_lds_b32 %1, %3, off"
            :: "v"(ldsB), "v"(ldsC), "v"(gB), "v"(gC) : "memory");

        float xv = xb[hd * HEADDIM + p];
        float dtv = dt[row * NHEADS + hd];
        float dav = dA[row * NHEADS + hd];

        asm volatile("s_wait_asynccnt 0x0" ::: "memory");
        __syncthreads();

        float dtx = dtv * xv;
        float acc = 0.f;
#pragma unroll
        for (int n = 0; n < D_STATE; ++n) {
            st[n] = st[n] * dav + dtx * Bs[n];
            acc  += st[n] * Cs[n];
        }
        y[(size_t)row * D_INNER + hd * HEADDIM + p] = acc + Dv * xv;
        __syncthreads();
    }
}

// ---------------------------------------------------------------------------
// y = y_scan * silu(z); RMS-norm over D_INNER; write bf16 for out_proj GEMM
// ---------------------------------------------------------------------------
__global__ void __launch_bounds__(256)
gate_rms_kernel(const float* __restrict__ zx,
                const float* __restrict__ y_scan,
                const float* __restrict__ rms_w,
                bf16_t* __restrict__ y_bf, int layer) {
    __shared__ float red[8];
    int row = blockIdx.x;
    float vals[4];
    float ss = 0.f;
#pragma unroll
    for (int j = 0; j < 4; ++j) {
        int i = j * 256 + threadIdx.x;
        float z  = zx[(size_t)row * D_IN_PROJ + i];
        float yv = y_scan[(size_t)row * D_INNER + i] * silu(z);
        vals[j] = yv;
        ss += yv * yv;
    }
    float tot = block_reduce_sum(ss, red);
    float scale = rsqrtf(tot * (1.0f / D_INNER) + 1e-5f);
#pragma unroll
    for (int j = 0; j < 4; ++j) {
        int i = j * 256 + threadIdx.x;
        y_bf[(size_t)row * D_INNER + i] =
            (bf16_t)(vals[j] * scale * rms_w[(size_t)layer * D_INNER + i]);
    }
}

// ---------------------------------------------------------------------------
// Final LayerNorm over D_MODEL
// ---------------------------------------------------------------------------
__global__ void __launch_bounds__(256)
ln_kernel(const float* __restrict__ h,
          const float* __restrict__ w,
          const float* __restrict__ bias,
          float* __restrict__ out) {
    __shared__ float red[8];
    int row = blockIdx.x;
    float v = h[(size_t)row * D_MODEL + threadIdx.x];
    float mu = block_reduce_sum(v, red) * (1.0f / D_MODEL);
    float d = v - mu;
    float var = block_reduce_sum(d * d, red) * (1.0f / D_MODEL);
    out[(size_t)row * D_MODEL + threadIdx.x] =
        d * rsqrtf(var + 1e-5f) * w[threadIdx.x] + bias[threadIdx.x];
}

// Mean-pool over sequence
__global__ void __launch_bounds__(256)
pool_kernel(const float* __restrict__ hln, float* __restrict__ pooled) {
    int b = blockIdx.x, d = threadIdx.x;
    float s = 0.f;
    for (int t = 0; t < L_SEQ; ++t)
        s += hln[((size_t)b * L_SEQ + t) * D_MODEL + d];
    pooled[b * D_MODEL + d] = s * (1.0f / L_SEQ);
}

// Regression head: gelu(pooled @ W1^T + b1) @ W2^T + b2
__global__ void __launch_bounds__(128)
head_kernel(const float* __restrict__ pooled,
            const float* __restrict__ w1, const float* __restrict__ b1,
            const float* __restrict__ w2, const float* __restrict__ b2,
            float* __restrict__ out) {
    __shared__ float g[128];
    int b = blockIdx.x, j = threadIdx.x;
    float acc = b1[j];
    const float* pp = pooled + b * D_MODEL;
    const float* wp = w1 + j * D_MODEL;
    for (int d = 0; d < D_MODEL; ++d) acc += pp[d] * wp[d];
    g[j] = gelu_exact(acc) * w2[j];
    __syncthreads();
    if (j == 0) {
        float s = b2[0];
        for (int i = 0; i < 128; ++i) s += g[i];
        out[b] = s;
    }
}

// Generic f32 -> bf16 convert (weights)
__global__ void f32_to_bf16_kernel(const float* __restrict__ in,
                                   bf16_t* __restrict__ out, int n) {
    int i = blockIdx.x * blockDim.x + threadIdx.x;
    if (i < n) out[i] = (bf16_t)in[i];
}

// ---------------------------------------------------------------------------
// Host launcher
// ---------------------------------------------------------------------------
static inline int cdiv(int a, int b) { return (a + b - 1) / b; }

extern "C" void kernel_launch(void* const* d_in, const int* in_sizes, int n_in,
                              void* d_out, int out_size, void* d_ws, size_t ws_size,
                              hipStream_t stream) {
    const float* x         = (const float*)d_in[0];
    const float* conv1_w   = (const float*)d_in[1];
    const float* conv1_b   = (const float*)d_in[2];
    const float* conv2_w   = (const float*)d_in[3];
    const float* conv2_b   = (const float*)d_in[4];
    const float* in_proj_w = (const float*)d_in[5];
    const float* dw_conv_w = (const float*)d_in[6];
    const float* dw_conv_b = (const float*)d_in[7];
    const float* dt_bias   = (const float*)d_in[8];
    const float* A_log     = (const float*)d_in[9];
    const float* ssm_D     = (const float*)d_in[10];
    const float* rms_w     = (const float*)d_in[11];
    const float* out_proj_w= (const float*)d_in[12];
    const float* ln_w      = (const float*)d_in[13];
    const float* ln_b      = (const float*)d_in[14];
    const float* reg1_w    = (const float*)d_in[15];
    const float* reg1_b    = (const float*)d_in[16];
    const float* reg2_w    = (const float*)d_in[17];
    const float* reg2_b    = (const float*)d_in[18];
    float* out = (float*)d_out;

    // ---- carve workspace -------------------------------------------------
    size_t off = 0;
    char* base = (char*)d_ws;
    auto carve = [&](size_t bytes) -> void* {
        void* p = base + off;
        off += (bytes + 255) & ~(size_t)255;
        return p;
    };
    float*  zx   = (float*) carve((size_t)ROWS * D_IN_PROJ * 4);  // zxbcdt
    float*  hf   = (float*) carve((size_t)ROWS * D_MODEL * 4);    // h (f32)
    bf16_t* hb   = (bf16_t*)carve((size_t)ROWS * D_MODEL * 2);    // h (bf16)
    float*  xBC  = (float*) carve((size_t)ROWS * CONV_DIM * 4);
    float*  dtb  = (float*) carve((size_t)ROWS * NHEADS * 4);
    float*  dAb  = (float*) carve((size_t)ROWS * NHEADS * 4);
    float*  ysc  = (float*) carve((size_t)ROWS * D_INNER * 4);
    bf16_t* ybf  = (bf16_t*)carve((size_t)ROWS * D_INNER * 2);
    bf16_t* wib  = (bf16_t*)carve((size_t)N_LAYERS * D_IN_PROJ * D_MODEL * 2);
    bf16_t* wob  = (bf16_t*)carve((size_t)N_LAYERS * D_MODEL * D_INNER * 2);
    float*  pooled = (float*)carve((size_t)BATCH * D_MODEL * 4);
    // temporal aliases (lifetimes don't overlap with zx usage):
    float*  h1     = zx;   // conv1 output, before layer loop
    float*  ln_out = zx;   // final LN output, after layer loop

    // ---- weight precision convert (bf16) --------------------------------
    {
        int n_in_w  = N_LAYERS * D_IN_PROJ * D_MODEL;
        int n_out_w = N_LAYERS * D_MODEL * D_INNER;
        f32_to_bf16_kernel<<<cdiv(n_in_w, 256), 256, 0, stream>>>(in_proj_w, wib, n_in_w);
        f32_to_bf16_kernel<<<cdiv(n_out_w, 256), 256, 0, stream>>>(out_proj_w, wob, n_out_w);
    }

    // ---- conv stem -------------------------------------------------------
    conv1_kernel<<<cdiv(BATCH * 64 * L1_LEN, 256), 256, 0, stream>>>(
        x, conv1_w, conv1_b, h1);
    conv2_kernel<<<cdiv(BATCH * D_MODEL * L_SEQ, 256), 256, 0, stream>>>(
        h1, conv2_w, conv2_b, hf, hb);

    // ---- Mamba2 layers ---------------------------------------------------
    const int wpb = 8; // waves per 256-thread block
    const int mtiles = ROWS / 16;                       // 512
    for (int l = 0; l < N_LAYERS; ++l) {
        const bf16_t* wi = wib + (size_t)l * D_IN_PROJ * D_MODEL;
        const bf16_t* wo = wob + (size_t)l * D_MODEL * D_INNER;

        // in_proj: zx[8192x2192] = h_bf16[8192x256] @ W^T
        // main: 34 groups of 64 cols (NSUB=4); tail: 1 group of 16 cols.
        gemm_bf16_wmma<4, false><<<cdiv(mtiles * 34, wpb), 256, 0, stream>>>(
            hb, wi, zx, nullptr, ROWS, D_IN_PROJ, D_MODEL, 0, 34);
        gemm_bf16_wmma<1, false><<<cdiv(mtiles * 1, wpb), 256, 0, stream>>>(
            hb, wi, zx, nullptr, ROWS, D_IN_PROJ, D_MODEL, 34 * 64, 1);

        dtda_kernel<<<cdiv(ROWS * NHEADS, 256), 256, 0, stream>>>(
            zx, dt_bias, A_log, dtb, dAb, l);
        dwconv_kernel<<<cdiv(ROWS * CONV_DIM, 256), 256, 0, stream>>>(
            zx, dw_conv_w, dw_conv_b, xBC, l);
        scan_kernel<<<BATCH * NHEADS, 64, 0, stream>>>(
            xBC, dtb, dAb, ssm_D, ysc, l);
        gate_rms_kernel<<<ROWS, 256, 0, stream>>>(zx, ysc, rms_w, ybf, l);

        // out_proj: h[8192x256] = y_bf16[8192x1024] @ W^T (+ fused bf16 copy)
        gemm_bf16_wmma<4, true><<<cdiv(mtiles * 4, wpb), 256, 0, stream>>>(
            ybf, wo, hf, hb, ROWS, D_MODEL, D_INNER, 0, 4);
    }

    // ---- tail: LN -> pool -> head ---------------------------------------
    ln_kernel<<<ROWS, 256, 0, stream>>>(hf, ln_w, ln_b, ln_out);
    pool_kernel<<<BATCH, 256, 0, stream>>>(ln_out, pooled);
    head_kernel<<<BATCH, 128, 0, stream>>>(pooled, reg1_w, reg1_b, reg2_w, reg2_b, out);

    (void)in_sizes; (void)n_in; (void)out_size; (void)ws_size;
}
